// IterativeDecoderOutput_80393197847212
// MI455X (gfx1250) — matrix operand order repaired and checked by reading, not compiled, not run on hardware
//
#include <hip/hip_runtime.h>
#include <hip/hip_bf16.h>
#include <math.h>

// Problem constants (match reference)
#define Bn 32
#define Sn 400
#define Mn 400
#define Hn 200
#define Pn 16
#define Tn 4
#define PH (Pn * Hn)        // 3200
#define Rn (Bn * Sn)        // 12800
#define K1P 416             // 400 padded to 13*32
#define K2P 224             // 200 padded to 7*32

typedef __attribute__((ext_vector_type(16))) __bf16 v16bf;
typedef __attribute__((ext_vector_type(8)))  __bf16 v8bf;
typedef __attribute__((ext_vector_type(8)))  float  v8f;

__device__ __forceinline__ void wait_async0() {
#if __has_builtin(__builtin_amdgcn_s_wait_asynccnt)
  __builtin_amdgcn_s_wait_asynccnt(0);
#else
  asm volatile("s_wait_asynccnt 0x0" ::: "memory");
#endif
}

// ---------------------------------------------------------------------------
// Fused WMMA GEMM + maxout:
//   Dst[row, h] = max_{p<P} ( sum_k A[row,k]*W[p*H + h, k] + bias[b(row)*bs + p*H + h] )
// A: [R, lda] bf16 (zero-padded K), W: [PH, ldw] bf16 (zero-padded K),
// bias: f32, stride bs = PH (per-batch) or 0 (shared), Dst: [R, ldd] bf16.
// One wave -> one 16(rows) x 16(h) tile; 8 waves/block share the h-tile, so
// W tiles are staged once per block into LDS with async global->LDS copies
// (ASYNCcnt), double-buffered across the P=16 maxout passes. A-fragments are
// cached in registers for the whole kernel (only W changes per pass).
// ---------------------------------------------------------------------------
template <int KSTEPS>
__global__ __launch_bounds__(256)
void wmma_maxout_kernel(const __bf16* __restrict__ A, int lda,
                        const __bf16* __restrict__ W, int ldw,
                        const float* __restrict__ bias, int bias_stride,
                        __bf16* __restrict__ Dst, int ldd) {
  constexpr int KE = KSTEPS * 32;          // K extent (elements)
  constexpr int PITCH = KE + 8;            // +16B row pad -> conflict-free ds reads
  constexpr int CPR = KSTEPS * 4;          // 16B chunks per W row
  constexpr int CHUNKS = 16 * CPR;         // chunks per staged tile
  __shared__ __bf16 wtile[2][16 * PITCH];

  const int tid = threadIdx.x;
  const int lane = tid & 31;
  const int wave = tid >> 5;
  const int rowTile = blockIdx.x * 8 + wave;       // 0..R/16-1
  const int row0 = rowTile * 16;
  const int h0 = blockIdx.y * 16;                  // h tile (13 tiles, last partial)
  const int b = row0 / Sn;                         // 16-row tiles never cross batch
  const int lrow = lane & 15;
  const int lhi  = lane >> 4;

  // Stage W rows {p*H + clamp(h0+i)} (i<16) into wtile[buf] via async LDS copies.
  auto stage = [&](int p, int buf) {
    for (int c = tid; c < CHUNKS; c += 256) {
      const int i = c / CPR;
      const int j = c - i * CPR;
      int hr = h0 + i; if (hr >= Hn) hr = Hn - 1;  // clamp (dup rows, never stored)
      const __bf16* gp = W + (size_t)(p * Hn + hr) * ldw + j * 8;
      const __bf16* lp = &wtile[buf][i * PITCH + j * 8];
      const uint32_t la = (uint32_t)(uintptr_t)lp;     // LDS byte offset (addr[31:0])
      asm volatile("global_load_async_to_lds_b128 %0, %1, off"
                   :: "v"(la), "v"((uint64_t)(uintptr_t)gp)
                   : "memory");
    }
  };

  stage(0, 0);   // prologue copy overlaps the A-fragment loads below

  // ---- load & cache all A fragments (16x32 bf16 per k-step) ----
  v16bf afrag[KSTEPS];
  const __bf16* arow = A + (size_t)(row0 + lrow) * lda + 8 * lhi;
#pragma unroll
  for (int ks = 0; ks < KSTEPS; ++ks) {
    v8bf lo = *(const v8bf*)(arow + ks * 32);
    v8bf hi = *(const v8bf*)(arow + ks * 32 + 16);
    v16bf af;
#pragma unroll
    for (int i = 0; i < 8; ++i) { af[i] = lo[i]; af[i + 8] = hi[i]; }
    afrag[ks] = af;
  }

  const int hcol = h0 + lrow;                      // output column (may be >=200)
  const int hc = hcol < Hn ? hcol : (Hn - 1);      // clamped for safe bias loads

  v8f vmax;
#pragma unroll
  for (int r = 0; r < 8; ++r) vmax[r] = -3.0e38f;

  wait_async0();
  __syncthreads();

  for (int p = 0; p < Pn; ++p) {
    const int buf = p & 1;
    if (p + 1 < Pn) stage(p + 1, buf ^ 1);         // prefetch next pass into other buffer

    // accumulator seeded with the r-dependent bias (same for all 16 rows)
    const float bv = bias[(size_t)b * bias_stride + p * Hn + hc];
    v8f c;
#pragma unroll
    for (int r = 0; r < 8; ++r) c[r] = bv;

    const __bf16* wrow = &wtile[buf][lrow * PITCH + 8 * lhi];
#pragma unroll
    for (int ks = 0; ks < KSTEPS; ++ks) {
      v8bf lo = *(const v8bf*)(wrow + ks * 32);
      v8bf hi = *(const v8bf*)(wrow + ks * 32 + 16);
      v16bf bfrag;
#pragma unroll
      for (int i = 0; i < 8; ++i) { bfrag[i] = lo[i]; bfrag[i + 8] = hi[i]; }
      c = __builtin_amdgcn_wmma_f32_16x16x32_bf16(
              false, afrag[ks], false, bfrag, (short)0, c, false, false);
    }
#pragma unroll
    for (int r = 0; r < 8; ++r) vmax[r] = fmaxf(vmax[r], c[r]);

    wait_async0();       // own async copies done ...
    __syncthreads();     // ... and everyone's writes visible before next pass
  }

  if (hcol < Hn) {
#pragma unroll
    for (int r = 0; r < 8; ++r) {
      const int m = r + 8 * lhi;                   // D layout: M = r (+8 for hi lanes)
      Dst[(size_t)(row0 + m) * ldd + hcol] = (__bf16)vmax[r];
    }
  }
}

// ---------------------------------------------------------------------------
// Small scalar kernels
// ---------------------------------------------------------------------------
__global__ void zero_f32_kernel(float* p, int n) {
  int i = blockIdx.x * blockDim.x + threadIdx.x;
  if (i < n) p[i] = 0.f;
}
__global__ void zero_bf16_kernel(__bf16* p, int n) {
  int i = blockIdx.x * blockDim.x + threadIdx.x;
  if (i < n) p[i] = (__bf16)0.f;
}

// f32 -> bf16 with row padding (pad columns set to zero for clean WMMA K loop)
__global__ void convert_pad_kernel(const float* __restrict__ src, int srow, int ncols,
                                   __bf16* __restrict__ dst, int drow, int nrows) {
  int idx = blockIdx.x * blockDim.x + threadIdx.x;
  int total = nrows * drow;
  if (idx >= total) return;
  int rr = idx / drow, cc = idx - rr * drow;
  float v = (cc < ncols) ? src[(size_t)rr * srow + cc] : 0.f;
  dst[idx] = (__bf16)v;
}

__global__ void init_state_kernel(const float* __restrict__ mask, int* sp, int* ep) {
  int b = threadIdx.x;
  if (b < Bn) {
    float s = 0.f;
    for (int i = 0; i < Sn; ++i) s += mask[b * Sn + i];
    sp[b] = 0;
    ep[b] = (int)s - 1;
  }
}

__device__ __forceinline__ float sigmoidf(float x) { return 1.f / (1.f + __expf(-x)); }

// gates g[b, 0:4H] = x @ Wih.T + bih + h @ Whh.T + bhh ; x = [mod[b,sp], mod[b,ep]]
__global__ void lstm_gates_kernel(const float* __restrict__ mod,
                                  const int* __restrict__ sp, const int* __restrict__ ep,
                                  const float* __restrict__ h,
                                  const float* __restrict__ Wih, const float* __restrict__ Whh,
                                  const float* __restrict__ bih, const float* __restrict__ bhh,
                                  float* __restrict__ g) {
  int idx = blockIdx.x * blockDim.x + threadIdx.x;
  if (idx >= Bn * 4 * Hn) return;
  int b = idx / (4 * Hn), j = idx - b * (4 * Hn);
  const float* us = mod + ((size_t)b * Sn + sp[b]) * Mn;
  const float* ue = mod + ((size_t)b * Sn + ep[b]) * Mn;
  const float* wr = Wih + (size_t)j * (2 * Mn);
  float acc = bih[j] + bhh[j];
  for (int k = 0; k < Mn; ++k) acc += us[k] * wr[k];
  for (int k = 0; k < Mn; ++k) acc += ue[k] * wr[Mn + k];
  const float* w2 = Whh + (size_t)j * Hn;
  const float* hb = h + b * Hn;
  for (int k = 0; k < Hn; ++k) acc += hb[k] * w2[k];
  g[idx] = acc;
}

__global__ void lstm_update_kernel(const float* __restrict__ g, float* h, float* c) {
  int idx = blockIdx.x * blockDim.x + threadIdx.x;
  if (idx >= Bn * Hn) return;
  int b = idx / Hn, j = idx - b * Hn;
  const float* gb = g + (size_t)b * 4 * Hn;
  float gi = sigmoidf(gb[j]);
  float gf = sigmoidf(gb[Hn + j]);
  float gg = tanhf(gb[2 * Hn + j]);
  float go = sigmoidf(gb[3 * Hn + j]);
  float c2 = gf * c[idx] + gi * gg;
  c[idx] = c2;
  h[idx] = go * tanhf(c2);
}

// r[b, j] = tanh( [h, us, ue] @ rW[j, :].T )
__global__ void r_kernel(const float* __restrict__ mod,
                         const int* __restrict__ sp, const int* __restrict__ ep,
                         const float* __restrict__ h, const float* __restrict__ rW,
                         float* __restrict__ r) {
  int idx = blockIdx.x * blockDim.x + threadIdx.x;
  if (idx >= Bn * Hn) return;
  int b = idx / Hn, j = idx - b * Hn;
  const float* w = rW + (size_t)j * (2 * Mn + Hn);
  const float* hb = h + b * Hn;
  float acc = 0.f;
  for (int k = 0; k < Hn; ++k) acc += hb[k] * w[k];
  const float* us = mod + ((size_t)b * Sn + sp[b]) * Mn;
  for (int k = 0; k < Mn; ++k) acc += us[k] * w[Hn + k];
  const float* ue = mod + ((size_t)b * Sn + ep[b]) * Mn;
  for (int k = 0; k < Mn; ++k) acc += ue[k] * w[Hn + Mn + k];
  r[idx] = tanhf(acc);
}

// rbias[b, j] = b1[j] + r[b,:] @ W1[j, M:].T   (the r-dependent part of x@W1.T)
__global__ void rbias_kernel(const float* __restrict__ r, const float* __restrict__ W1,
                             const float* __restrict__ b1, float* __restrict__ rbias) {
  int idx = blockIdx.x * blockDim.x + threadIdx.x;
  if (idx >= Bn * PH) return;
  int b = idx / PH, j = idx - b * PH;
  const float* w = W1 + (size_t)j * (Mn + Hn) + Mn;
  const float* rb = r + b * Hn;
  float acc = b1[j];
  for (int k = 0; k < Hn; ++k) acc += rb[k] * w[k];
  rbias[idx] = acc;
}

// logits[row] = max_q ( b3[q] + sum_h m1[row,h]*W3[q,h] + m2[row,h]*W3[q,H+h] ), masked
__global__ void logits_kernel(const __bf16* __restrict__ m1, const __bf16* __restrict__ m2,
                              const float* __restrict__ W3, const float* __restrict__ b3,
                              const float* __restrict__ mask, float* __restrict__ logits) {
  int row = blockIdx.x * blockDim.x + threadIdx.x;
  if (row >= Rn) return;
  int b = row / Sn, s = row - b * Sn;
  const __bf16* a1 = m1 + (size_t)row * K2P;
  const __bf16* a2 = m2 + (size_t)row * K2P;
  float best = -3.0e38f;
  for (int q = 0; q < Pn; ++q) {
    const float* w = W3 + (size_t)q * (2 * Hn);
    float acc = b3[q];
    for (int hh = 0; hh < Hn; ++hh)
      acc += (float)a1[hh] * w[hh] + (float)a2[hh] * w[Hn + hh];
    best = fmaxf(best, acc);
  }
  logits[row] = (mask[b * Sn + s] > 0.f) ? best : -1.0e30f;
}

// per-batch log_softmax over S + argmax -> prev[b]; out[(b*T + t)*S + s]
__global__ void softmax_kernel(const float* __restrict__ logits,
                               float* __restrict__ out, int* __restrict__ prev, int t) {
  __shared__ float smax[128];
  __shared__ int   sidx[128];
  __shared__ float ssum[128];
  const int b = blockIdx.x, tid = threadIdx.x;
  float mv = -3.0e38f; int mi = Sn;
  for (int s = tid; s < Sn; s += 128) {
    float v = logits[b * Sn + s];
    if (v > mv) { mv = v; mi = s; }
  }
  smax[tid] = mv; sidx[tid] = mi;
  __syncthreads();
  for (int off = 64; off > 0; off >>= 1) {
    if (tid < off) {
      if (smax[tid + off] > smax[tid] ||
          (smax[tid + off] == smax[tid] && sidx[tid + off] < sidx[tid])) {
        smax[tid] = smax[tid + off]; sidx[tid] = sidx[tid + off];
      }
    }
    __syncthreads();
  }
  const float gmax = smax[0];
  if (tid == 0) prev[b] = sidx[0];
  float ls = 0.f;
  for (int s = tid; s < Sn; s += 128) ls += __expf(logits[b * Sn + s] - gmax);
  ssum[tid] = ls;
  __syncthreads();
  for (int off = 64; off > 0; off >>= 1) {
    if (tid < off) ssum[tid] += ssum[tid + off];
    __syncthreads();
  }
  const float lse = logf(ssum[0]);
  for (int s = tid; s < Sn; s += 128)
    out[((size_t)b * Tn + t) * Sn + s] = logits[b * Sn + s] - gmax - lse;
}

// ---------------------------------------------------------------------------
extern "C" void kernel_launch(void* const* d_in, const int* in_sizes, int n_in,
                              void* d_out, int out_size, void* d_ws, size_t ws_size,
                              hipStream_t stream) {
  (void)in_sizes; (void)n_in; (void)out_size; (void)ws_size;
  const float* mod   = (const float*)d_in[1];
  const float* mask  = (const float*)d_in[2];
  const float* Wih   = (const float*)d_in[3];
  const float* Whh   = (const float*)d_in[4];
  const float* bih   = (const float*)d_in[5];
  const float* bhh   = (const float*)d_in[6];
  const float* rWp[2]  = { (const float*)d_in[7],  (const float*)d_in[14] };
  const float* W1p[2]  = { (const float*)d_in[8],  (const float*)d_in[15] };
  const float* b1p[2]  = { (const float*)d_in[9],  (const float*)d_in[16] };
  const float* W2p[2]  = { (const float*)d_in[10], (const float*)d_in[17] };
  const float* b2p[2]  = { (const float*)d_in[11], (const float*)d_in[18] };
  const float* W3p[2]  = { (const float*)d_in[12], (const float*)d_in[19] };
  const float* b3p[2]  = { (const float*)d_in[13], (const float*)d_in[20] };
  float* out = (float*)d_out;

  // ---- workspace carve-out (256B aligned) ----
  char* ws = (char*)d_ws;
  size_t off = 0;
  auto carve = [&](size_t bytes) -> char* {
    char* p = ws + off;
    off = (off + bytes + 255) & ~(size_t)255;
    return p;
  };
  __bf16* mod_bf = (__bf16*)carve((size_t)Rn * K1P * 2);
  __bf16* w1bf[2] = { (__bf16*)carve((size_t)PH * K1P * 2), (__bf16*)carve((size_t)PH * K1P * 2) };
  __bf16* w2bf[2] = { (__bf16*)carve((size_t)PH * K2P * 2), (__bf16*)carve((size_t)PH * K2P * 2) };
  __bf16* m1bf = (__bf16*)carve((size_t)Rn * K2P * 2);
  __bf16* m2bf = (__bf16*)carve((size_t)Rn * K2P * 2);
  float* gbuf   = (float*)carve((size_t)Bn * 4 * Hn * 4);
  float* hbuf   = (float*)carve((size_t)Bn * Hn * 4);
  float* cbuf   = (float*)carve((size_t)Bn * Hn * 4);
  float* rbuf   = (float*)carve((size_t)Bn * Hn * 4);
  float* rbias  = (float*)carve((size_t)Bn * PH * 4);
  float* logits = (float*)carve((size_t)Rn * 4);
  int*   sp     = (int*)carve(Bn * 4);
  int*   ep     = (int*)carve(Bn * 4);

  const int TB = 256;
  auto nb = [](int n, int t) { return (n + t - 1) / t; };

  // ---- one-time staging (deterministic: rerun every call) ----
  zero_f32_kernel<<<nb(Bn * Hn, TB), TB, 0, stream>>>(hbuf, Bn * Hn);
  zero_f32_kernel<<<nb(Bn * Hn, TB), TB, 0, stream>>>(cbuf, Bn * Hn);
  zero_bf16_kernel<<<nb(Rn * K2P, TB), TB, 0, stream>>>(m1bf, Rn * K2P);
  zero_bf16_kernel<<<nb(Rn * K2P, TB), TB, 0, stream>>>(m2bf, Rn * K2P);
  init_state_kernel<<<1, 32, 0, stream>>>(mask, sp, ep);
  convert_pad_kernel<<<nb(Rn * K1P, TB), TB, 0, stream>>>(mod, Mn, Mn, mod_bf, K1P, Rn);
  for (int i = 0; i < 2; ++i) {
    convert_pad_kernel<<<nb(PH * K1P, TB), TB, 0, stream>>>(W1p[i], Mn + Hn, Mn, w1bf[i], K1P, PH);
    convert_pad_kernel<<<nb(PH * K2P, TB), TB, 0, stream>>>(W2p[i], Hn, Hn, w2bf[i], K2P, PH);
  }

  dim3 wgrid(Rn / 16 / 8, (Hn + 15) / 16);  // 100 x 13, 8 waves/block, 1 tile/wave

  for (int t = 0; t < Tn; ++t) {
    // LSTM cell (gathers u_s/u_e via current prev indices)
    lstm_gates_kernel<<<nb(Bn * 4 * Hn, TB), TB, 0, stream>>>(mod, sp, ep, hbuf,
                                                              Wih, Whh, bih, bhh, gbuf);
    lstm_update_kernel<<<nb(Bn * Hn, TB), TB, 0, stream>>>(gbuf, hbuf, cbuf);

    for (int head = 0; head < 2; ++head) {  // 0 = start head, 1 = end head
      r_kernel<<<nb(Bn * Hn, TB), TB, 0, stream>>>(mod, sp, ep, hbuf, rWp[head], rbuf);
      rbias_kernel<<<nb(Bn * PH, TB), TB, 0, stream>>>(rbuf, W1p[head], b1p[head], rbias);
      // m1 = maxout_P( mod @ W1mod.T + rbias )   [WMMA, K=416 -> 13 steps]
      wmma_maxout_kernel<13><<<wgrid, TB, 0, stream>>>(mod_bf, K1P, w1bf[head], K1P,
                                                       rbias, PH, m1bf, K2P);
      // m2 = maxout_P( m1 @ W2.T + b2 )          [WMMA, K=224 -> 7 steps]
      wmma_maxout_kernel<7><<<wgrid, TB, 0, stream>>>(m1bf, K2P, w2bf[head], K2P,
                                                      b2p[head], 0, m2bf, K2P);
      logits_kernel<<<nb(Rn, TB), TB, 0, stream>>>(m1bf, m2bf, W3p[head], b3p[head],
                                                   mask, logits);
      float* obase = out + (size_t)head * Bn * Tn * Sn;  // lp1 then lp2, concatenated
      int* prev = (head == 0) ? sp : ep;                 // argmax feeds next gather
      softmax_kernel<<<Bn, 128, 0, stream>>>(logits, obase, prev, t);
    }
  }
}
